// CACOM_Agent_10557029614080
// MI455X (gfx1250) — matrix operand order repaired and checked by reading, not compiled, not run on hardware
//
#include <hip/hip_runtime.h>
#include <hip/hip_bf16.h>
#include <math.h>

typedef _Float16 h16;
typedef __attribute__((ext_vector_type(16))) _Float16 v16h;
typedef __attribute__((ext_vector_type(8)))  float    v8f;

#define SCALE_ATT 0.08838834764831845f  // 1/sqrt(128)

__device__ __forceinline__ int lane_id() { return (int)(threadIdx.x & 31u); }

__device__ __forceinline__ v8f zero8() {
  v8f c;
#pragma unroll
  for (int i = 0; i < 8; ++i) c[i] = 0.f;
  return c;
}

__device__ __forceinline__ v8f bias8(const float* __restrict__ b) {
  float bv = b[lane_id() & 15];
  v8f c;
#pragma unroll
  for (int i = 0; i < 8; ++i) c[i] = bv;
  return c;
}

__device__ __forceinline__ v8f wmma_f16(v16h a, v16h b, v8f c) {
  return __builtin_amdgcn_wmma_f32_16x16x32_f16(false, a, false, b, (short)0, c, false, false);
}

// ---- WMMA fragment loaders per CDNA5 ISA layouts (wave32) ----
// A: 16(M) x 32(K), row-major source. lane = row (l&15); VGPR i holds K pair
// K = (i>>2)*16 + (lane>=16 ? 8 : 0) + (i&3)*2
__device__ __forceinline__ v16h ldA(const h16* __restrict__ src, int ld) {
  int l = lane_id(), r = l & 15, g = l >> 4;
  const h16* p = src + r * ld + (g << 3);
  v16h a;
#pragma unroll
  for (int i = 0; i < 8; ++i) {
    int k = ((i >> 2) << 4) + ((i & 3) << 1);
    a[2 * i] = p[k];
    a[2 * i + 1] = p[k + 1];
  }
  return a;
}

// A with row guard: rows >= nrows are zero (for 15-row resp tiles)
__device__ __forceinline__ v16h ldA_g(const h16* __restrict__ src, int ld, int nrows) {
  int l = lane_id(), r = l & 15, g = l >> 4;
  v16h a;
  if (r < nrows) {
    const h16* p = src + r * ld + (g << 3);
#pragma unroll
    for (int i = 0; i < 8; ++i) {
      int k = ((i >> 2) << 4) + ((i & 3) << 1);
      a[2 * i] = p[k];
      a[2 * i + 1] = p[k + 1];
    }
  } else {
#pragma unroll
    for (int i = 0; i < 16; ++i) a[i] = (h16)0.f;
  }
  return a;
}

// B (32K x 16N), "weight style": B[k][n] = W[n][k], W row-major with stride ldw.
// lane = col n (l&15); element s: K = (s<8 ? s : s+8) + (lane>=16 ? 8 : 0)
__device__ __forceinline__ v16h ldBw(const h16* __restrict__ W, int ldw) {
  int l = lane_id(), n = l & 15, g = l >> 4;
  const h16* p = W + n * ldw + (g << 3);
  v16h b;
#pragma unroll
  for (int i = 0; i < 16; ++i) b[i] = p[i < 8 ? i : i + 8];
  return b;
}

// B (32K x 16N), "matrix style": B[k][n] = M[k][n], row-major stride ld (strided per lane)
__device__ __forceinline__ v16h ldBm(const h16* __restrict__ Msrc, int ld) {
  int l = lane_id(), n = l & 15, g = l >> 4;
  const h16* p = Msrc + (g << 3) * ld + n;
  v16h b;
#pragma unroll
  for (int i = 0; i < 16; ++i) b[i] = p[(i < 8 ? i : i + 8) * ld];
  return b;
}

// C/D (16x16 f32): lane = col n (l&15); VGPR i = row (lane>=16 ? 8 : 0) + i
__device__ __forceinline__ void stC_h(h16* __restrict__ dst, int ld, v8f c) {
  int l = lane_id(), n = l & 15, g = l >> 4;
#pragma unroll
  for (int i = 0; i < 8; ++i) dst[(g * 8 + i) * ld + n] = (h16)c[i];
}

__device__ __forceinline__ void stC_f(float* __restrict__ dst, int ld, v8f c) {
  int l = lane_id(), n = l & 15, g = l >> 4;
#pragma unroll
  for (int i = 0; i < 8; ++i) dst[(g * 8 + i) * ld + n] = c[i];
}

// row reductions across the 16 columns of a C-layout tile (lanes within a half-wave)
__device__ __forceinline__ float rmax16(float v) {
#pragma unroll
  for (int m = 1; m < 16; m <<= 1) v = fmaxf(v, __shfl_xor(v, m, 32));
  return v;
}
__device__ __forceinline__ float rsum16(float v) {
#pragma unroll
  for (int m = 1; m < 16; m <<= 1) v += __shfl_xor(v, m, 32);
  return v;
}

// =====================================================================
// f32 -> f16 conversion
__global__ void cvt_f2h(const float* __restrict__ src, h16* __restrict__ dst, long n) {
  long i = (long)blockIdx.x * blockDim.x + threadIdx.x;
  if (i < n) dst[i] = (h16)src[i];
}

// =====================================================================
// GEMM, A-fragments preloaded (K = KC*32 <= 256, compile-time). One wave per
// 16-row M tile; A read once. Optional z-batch (encoder segments) via bsA/bsC.
template <int KC, int CF32>
__global__ __launch_bounds__(256) void gemm_npre(
    const h16* __restrict__ A, long lda, long bsA,
    const h16* __restrict__ W, int ldw,
    const float* __restrict__ bias,
    void* __restrict__ Cv, long ldc, long bsC,
    int M, int N) {
  int wave = threadIdx.x >> 5;
  int mt = blockIdx.x * (blockDim.x >> 5) + wave;
  if (mt * 16 >= M) return;
  long z = blockIdx.y;
  const h16* Ab = A + z * bsA + (long)mt * 16 * lda;
  v16h af[KC];
#pragma unroll
  for (int kc = 0; kc < KC; ++kc) af[kc] = ldA(Ab + kc * 32, (int)lda);
  int ntc = N >> 4;
  for (int nt = 0; nt < ntc; ++nt) {
    v8f c = bias8(bias + nt * 16);
    const h16* Wn = W + (long)nt * 16 * ldw;
#pragma unroll
    for (int kc = 0; kc < KC; ++kc) c = wmma_f16(af[kc], ldBw(Wn + kc * 32, ldw), c);
    if (CF32) stC_f((float*)Cv + z * bsC + (long)mt * 16 * ldc + nt * 16, (int)ldc, c);
    else      stC_h((h16*)Cv + z * bsC + (long)mt * 16 * ldc + nt * 16, (int)ldc, c);
  }
}

// =====================================================================
// GEMM, k-outer with NT resident accumulator tiles (N = NT*16). A streamed once;
// weights re-read per m-tile but stay in L2. Dual A source via two clean k-loops
// (concat([hidden, feat]) for req).
template <int NT>
__global__ __launch_bounds__(256) void gemm_kouter(
    const h16* __restrict__ A, long lda,
    const h16* __restrict__ A2, long lda2, int K1, int K,
    const h16* __restrict__ W, int ldw,
    const float* __restrict__ bias,
    h16* __restrict__ C, long ldc, int M) {
  int wave = threadIdx.x >> 5;
  int mt = blockIdx.x * (blockDim.x >> 5) + wave;
  if (mt * 16 >= M) return;
  const h16* Ab = A + (long)mt * 16 * lda;
  const h16* A2b = A2 + (long)mt * 16 * lda2;
  v8f acc[NT];
#pragma unroll
  for (int nt = 0; nt < NT; ++nt) acc[nt] = bias8(bias + nt * 16);
  for (int k = 0; k < K1; k += 32) {
    v16h a = ldA(Ab + k, (int)lda);
#pragma unroll
    for (int nt = 0; nt < NT; ++nt)
      acc[nt] = wmma_f16(a, ldBw(W + (long)nt * 16 * ldw + k, ldw), acc[nt]);
  }
  for (int k = K1; k < K; k += 32) {
    v16h a = ldA(A2b + (k - K1), (int)lda2);
#pragma unroll
    for (int nt = 0; nt < NT; ++nt)
      acc[nt] = wmma_f16(a, ldBw(W + (long)nt * 16 * ldw + k, ldw), acc[nt]);
  }
#pragma unroll
  for (int nt = 0; nt < NT; ++nt)
    stC_h(C + (long)mt * 16 * ldc + nt * 16, (int)ldc, acc[nt]);
}

// =====================================================================
// Encoder block, fused per (b,a): kqv -> softmax(QK^T)V -> +obs -> FFN(+res) -> feat
#define EB_W 4
#define EB_PW (32 * 384 + 16 * 32 + 16 * 128)  // halfs per wave
__global__ __launch_bounds__(128) void enc_block(
    const h16* __restrict__ obs, const h16* __restrict__ Wkqv, const float* __restrict__ bkqv,
    const h16* __restrict__ Wf1, const float* __restrict__ bf1,
    const h16* __restrict__ Wf2, const float* __restrict__ bf2,
    h16* __restrict__ feat) {
  extern __shared__ h16 sm[];
  int wave = threadIdx.x >> 5;
  int ba = blockIdx.x * EB_W + wave;
  h16* kqv = sm + wave * EB_PW;  // 32 x 384: k | q | v ; rows 16..31 zero pad
  h16* P = kqv + 32 * 384;       // 16 x 32 probs (cols 16..31 zero)
  h16* ft = P + 16 * 32;         // 16 x 128 pre-FFN feat
  int l = lane_id();
  for (int i = l; i < 16 * 384; i += 32) kqv[16 * 384 + i] = (h16)0.f;
  for (int i = l; i < 256; i += 32) P[(i >> 4) * 32 + 16 + (i & 15)] = (h16)0.f;

  const h16* ob = obs + (long)ba * 2048;
  v16h aO[4];
#pragma unroll
  for (int kc = 0; kc < 4; ++kc) aO[kc] = ldA(ob + kc * 32, 128);
  for (int nt = 0; nt < 24; ++nt) {  // kqv = obs @ Wkqv^T + b
    v8f c = bias8(bkqv + nt * 16);
#pragma unroll
    for (int kc = 0; kc < 4; ++kc)
      c = wmma_f16(aO[kc], ldBw(Wkqv + nt * 16 * 128 + kc * 32, 128), c);
    stC_h(kqv + nt * 16, 384, c);
  }
  __syncthreads();
  // scores[s][j] = q[s].k[j]
  v8f s = zero8();
#pragma unroll
  for (int kc = 0; kc < 4; ++kc)
    s = wmma_f16(ldA(kqv + 128 + kc * 32, 384), ldBw(kqv + kc * 32, 384), s);
  v8f p;
#pragma unroll
  for (int i = 0; i < 8; ++i) {
    float v = s[i] * SCALE_ATT;
    float m = rmax16(v);
    float e = __expf(v - m);
    p[i] = e / rsum16(e);
  }
  stC_h(P, 32, p);
  __syncthreads();
  // ft = P @ V + obs
  v16h aP = ldA(P, 32);
  for (int et = 0; et < 8; ++et) {
    v8f c = zero8();
    c = wmma_f16(aP, ldBm(kqv + 256 + et * 16, 384), c);
    int n = l & 15, g = l >> 4;
#pragma unroll
    for (int i = 0; i < 8; ++i) c[i] += (float)ob[(g * 8 + i) * 128 + et * 16 + n];
    stC_h(ft + et * 16, 128, c);
  }
  __syncthreads();
  // t1 = lrelu(ft @ Wf1^T + b), stored into kqv rows 0..15 (stride 384, v no longer needed)
  v16h aF[4];
#pragma unroll
  for (int kc = 0; kc < 4; ++kc) aF[kc] = ldA(ft + kc * 32, 128);
  for (int nt = 0; nt < 16; ++nt) {
    v8f c = bias8(bf1 + nt * 16);
#pragma unroll
    for (int kc = 0; kc < 4; ++kc)
      c = wmma_f16(aF[kc], ldBw(Wf1 + nt * 16 * 128 + kc * 32, 128), c);
#pragma unroll
    for (int i = 0; i < 8; ++i) c[i] = c[i] >= 0.f ? c[i] : 0.01f * c[i];
    stC_h(kqv + nt * 16, 384, c);
  }
  __syncthreads();
  // feat = ft + t1 @ Wf2^T + b
  h16* fo = feat + (long)ba * 2048;
  for (int nt = 0; nt < 8; ++nt) {
    v8f c = bias8(bf2 + nt * 16);
#pragma unroll
    for (int kc = 0; kc < 8; ++kc)
      c = wmma_f16(ldA(kqv + kc * 32, 384), ldBw(Wf2 + nt * 16 * 256 + kc * 32, 256), c);
    int n = l & 15, g = l >> 4;
#pragma unroll
    for (int i = 0; i < 8; ++i) c[i] += (float)ft[(g * 8 + i) * 128 + nt * 16 + n];
    stC_h(fo + nt * 16, 128, c);
  }
}

// =====================================================================
// Response block, fused per (b,a): rkv -> softmax over s -> P'@rv -> rfc1/rfc2 -> mask -> offdiag scatter
#define RK_W 4
#define RK_PW (32 * 256 + 16 * 32 + 16 * 128)
__global__ __launch_bounds__(128) void resp_block(
    const h16* __restrict__ feat, const h16* __restrict__ rq2,
    const h16* __restrict__ Wrkv, const float* __restrict__ brkv,
    const h16* __restrict__ Wr1, const float* __restrict__ br1,
    const h16* __restrict__ Wr2, const float* __restrict__ br2,
    const float* __restrict__ mask, h16* __restrict__ resp_g) {
  extern __shared__ h16 sm[];
  int wave = threadIdx.x >> 5;
  int ba = blockIdx.x * RK_W + wave;
  int b = ba >> 4, a = ba & 15;
  h16* kv = sm + wave * RK_PW;  // 32 x 256: rk | rv; rows 16..31 zero
  h16* P = kv + 32 * 256;       // 16 x 32 (cols 16..31 zero)
  h16* rs = P + 16 * 32;        // 16 x 128
  int l = lane_id();
  for (int i = l; i < 16 * 256; i += 32) kv[16 * 256 + i] = (h16)0.f;
  for (int i = l; i < 256; i += 32) P[(i >> 4) * 32 + 16 + (i & 15)] = (h16)0.f;

  const h16* fb = feat + (long)ba * 2048;
  v16h aF[4];
#pragma unroll
  for (int kc = 0; kc < 4; ++kc) aF[kc] = ldA(fb + kc * 32, 128);
  for (int nt = 0; nt < 16; ++nt) {
    v8f c = bias8(brkv + nt * 16);
#pragma unroll
    for (int kc = 0; kc < 4; ++kc)
      c = wmma_f16(aF[kc], ldBw(Wrkv + nt * 16 * 128 + kc * 32, 128), c);
    stC_h(kv + nt * 16, 256, c);
  }
  __syncthreads();
  // scores'[j][s] = rq2[b,j] . rk[s]; softmax over s (columns)
  const h16* rqb = rq2 + (long)b * 16 * 128;
  v8f s = zero8();
#pragma unroll
  for (int kc = 0; kc < 4; ++kc)
    s = wmma_f16(ldA(rqb + kc * 32, 128), ldBw(kv + kc * 32, 256), s);
  v8f p;
#pragma unroll
  for (int i = 0; i < 8; ++i) {
    float v = s[i] * SCALE_ATT;
    float m = rmax16(v);
    float e = __expf(v - m);
    p[i] = e / rsum16(e);
  }
  stC_h(P, 32, p);
  __syncthreads();
  v16h aP = ldA(P, 32);
  for (int et = 0; et < 8; ++et) {  // resp[j] = P' @ rv
    v8f c = zero8();
    c = wmma_f16(aP, ldBm(kv + 128 + et * 16, 256), c);
    stC_h(rs + et * 16, 128, c);
  }
  __syncthreads();
  v16h aR[4];
#pragma unroll
  for (int kc = 0; kc < 4; ++kc) aR[kc] = ldA(rs + kc * 32, 128);
  for (int nt = 0; nt < 16; ++nt) {  // t1 = lrelu(resp @ Wr1^T + b) into kv rows 0..15
    v8f c = bias8(br1 + nt * 16);
#pragma unroll
    for (int kc = 0; kc < 4; ++kc)
      c = wmma_f16(aR[kc], ldBw(Wr1 + nt * 16 * 128 + kc * 32, 128), c);
#pragma unroll
    for (int i = 0; i < 8; ++i) c[i] = c[i] >= 0.f ? c[i] : 0.01f * c[i];
    stC_h(kv + nt * 16, 256, c);
  }
  __syncthreads();
  for (int nt = 0; nt < 4; ++nt) {  // rfc2 (N=64) + mask + offdiag scatter
    v8f c = bias8(br2 + nt * 16);
#pragma unroll
    for (int kc = 0; kc < 8; ++kc)
      c = wmma_f16(ldA(kv + kc * 32, 256), ldBw(Wr2 + nt * 16 * 256 + kc * 32, 256), c);
    int n = l & 15, g = l >> 4;
#pragma unroll
    for (int i = 0; i < 8; ++i) {
      int j = g * 8 + i;
      float v = c[i] * mask[(ba << 4) + j];
      if (j != a) {
        int m2 = a - (a > j ? 1 : 0);  // position of a in OFFDIAG[j]
        resp_g[((long)(b * 16 + j) * 15 + m2) * 64 + nt * 16 + n] = (h16)v;
      }
    }
  }
}

// =====================================================================
// Final attention, fused per (b,a): [fkqv(lrelu feat); mkqv(resp)] -> masked 32-token softmax -> P@V (+feat) -> xx
#define FA_W 4
#define FA_PW (32 * 384 + 32 * 32)
__global__ __launch_bounds__(128) void fattn_block(
    const h16* __restrict__ feat, const h16* __restrict__ resp_g,
    const h16* __restrict__ Wf, const float* __restrict__ bf,
    const h16* __restrict__ Wm, const float* __restrict__ bm,
    h16* __restrict__ xx) {
  extern __shared__ h16 sm[];
  int wave = threadIdx.x >> 5;
  int ba = blockIdx.x * FA_W + wave;
  h16* kqv = sm + wave * FA_PW;  // 32 x 384: rows 0..15 from feat, 16..30 from resp, 31 padded
  h16* P = kqv + 32 * 384;       // 32 x 32
  int l = lane_id();

  const h16* fb = feat + (long)ba * 2048;
  v16h aL[4];
#pragma unroll
  for (int kc = 0; kc < 4; ++kc) {
    v16h t = ldA(fb + kc * 32, 128);
#pragma unroll
    for (int e = 0; e < 16; ++e) {
      float xv = (float)t[e];
      t[e] = (h16)(xv >= 0.f ? xv : 0.01f * xv);
    }
    aL[kc] = t;
  }
  for (int nt = 0; nt < 24; ++nt) {
    v8f c = bias8(bf + nt * 16);
#pragma unroll
    for (int kc = 0; kc < 4; ++kc)
      c = wmma_f16(aL[kc], ldBw(Wf + nt * 16 * 128 + kc * 32, 128), c);
    stC_h(kqv + nt * 16, 384, c);
  }
  const h16* rg = resp_g + (long)ba * 15 * 64;
  v16h aM[2];
#pragma unroll
  for (int kc = 0; kc < 2; ++kc) aM[kc] = ldA_g(rg + kc * 32, 64, 15);
  for (int nt = 0; nt < 24; ++nt) {
    v8f c = bias8(bm + nt * 16);
#pragma unroll
    for (int kc = 0; kc < 2; ++kc)
      c = wmma_f16(aM[kc], ldBw(Wm + nt * 16 * 64 + kc * 32, 64), c);
    stC_h(kqv + 16 * 384 + nt * 16, 384, c);
  }
  __syncthreads();
  for (int mt = 0; mt < 2; ++mt) {
    v16h aQ[4];
#pragma unroll
    for (int kc = 0; kc < 4; ++kc) aQ[kc] = ldA(kqv + mt * 16 * 384 + 128 + kc * 32, 384);
    v8f s0 = zero8(), s1 = zero8();
#pragma unroll
    for (int kc = 0; kc < 4; ++kc) {
      s0 = wmma_f16(aQ[kc], ldBw(kqv + kc * 32, 384), s0);
      s1 = wmma_f16(aQ[kc], ldBw(kqv + 16 * 384 + kc * 32, 384), s1);
    }
    int n = l & 15;
    v8f p0, p1;
#pragma unroll
    for (int i = 0; i < 8; ++i) {
      float v0 = s0[i] * SCALE_ATT;
      float v1 = (n == 15) ? -1e30f : s1[i] * SCALE_ATT;  // mask padded token 31
      float m = fmaxf(rmax16(v0), rmax16(v1));
      float e0 = __expf(v0 - m), e1 = __expf(v1 - m);
      float sum = rsum16(e0) + rsum16(e1);
      p0[i] = e0 / sum;
      p1[i] = e1 / sum;
    }
    stC_h(P + mt * 16 * 32, 32, p0);
    stC_h(P + mt * 16 * 32 + 16, 32, p1);
  }
  __syncthreads();
  h16* xo = xx + (long)ba * 32 * 128;
  for (int mt = 0; mt < 2; ++mt) {
    v16h aP = ldA(P + mt * 16 * 32, 32);  // K=32 exact
    for (int et = 0; et < 8; ++et) {
      v8f c = zero8();
      c = wmma_f16(aP, ldBm(kqv + 256 + et * 16, 384), c);
      if (mt == 0) {
        int n = l & 15, g = l >> 4;
#pragma unroll
        for (int i = 0; i < 8; ++i) c[i] += (float)fb[(g * 8 + i) * 128 + et * 16 + n];
      }
      stC_h(xo + (long)mt * 16 * 128 + et * 16, 128, c);
    }
  }
}

// =====================================================================
// GRU pointwise
__global__ void gru_pw(const float* __restrict__ gi, const float* __restrict__ gh,
                       const float* __restrict__ h0, h16* __restrict__ hOut, long n) {
  long i = (long)blockIdx.x * blockDim.x + threadIdx.x;
  if (i >= n) return;
  long row = i >> 8;
  int c = (int)(i & 255);
  const float* gir = gi + row * 768;
  const float* ghr = gh + row * 768;
  float r = 1.f / (1.f + __expf(-(gir[c] + ghr[c])));
  float z = 1.f / (1.f + __expf(-(gir[256 + c] + ghr[256 + c])));
  float nn = tanhf(gir[512 + c] + r * ghr[512 + c]);
  float h = (1.f - z) * nn + z * h0[i];
  hOut[i] = (h16)h;
}

// =====================================================================
extern "C" void kernel_launch(void* const* d_in, const int* in_sizes, int n_in,
                              void* d_out, int out_size, void* d_ws, size_t ws_size,
                              hipStream_t stream) {
  (void)in_sizes; (void)n_in; (void)out_size; (void)ws_size;
  char* base = (char*)d_ws;
  size_t off = 0;
  auto alloc_h = [&](long nelem) -> h16* {
    h16* p = (h16*)(base + off);
    off = (off + (size_t)nelem * 2 + 255) & ~(size_t)255;
    return p;
  };
  auto alloc_f = [&](long nelem) -> float* {
    float* p = (float*)(base + off);
    off = (off + (size_t)nelem * 4 + 255) & ~(size_t)255;
    return p;
  };

  h16* x16 = alloc_h(8192L * 544);
  h16* h16b = alloc_h(8192L * 256);
  h16* wenc0 = alloc_h(128 * 64);
  h16* wenc1 = alloc_h(128 * 32);
  h16* winkqv = alloc_h(384 * 128);
  h16* winf1 = alloc_h(256 * 128);
  h16* winf2 = alloc_h(128 * 256);
  h16* wreq = alloc_h(64 * 2304);
  h16* wrkv = alloc_h(256 * 128);
  h16* wrq = alloc_h(128 * 64);
  h16* wrf1 = alloc_h(256 * 128);
  h16* wrf2 = alloc_h(64 * 256);
  h16* wafk = alloc_h(384 * 128);
  h16* wamk = alloc_h(384 * 64);
  h16* walin = alloc_h(256 * 3968);
  h16* wwih = alloc_h(768 * 256);
  h16* wwhh = alloc_h(768 * 256);
  h16* walin2 = alloc_h(32 * 256);
  h16* obs = alloc_h(131072L * 128);
  h16* feat = alloc_h(131072L * 128);
  h16* req16 = alloc_h(8192L * 64);
  h16* rq2 = alloc_h(8192L * 128);
  h16* respg = alloc_h(8192L * 15 * 64);
  h16* xx = alloc_h(8192L * 32 * 128);
  h16* xl16 = alloc_h(8192L * 256);
  h16* hnew = alloc_h(8192L * 256);
  float* gi = alloc_f(8192L * 768);
  float* gh = alloc_f(8192L * 768);

  // ---- f32 -> f16 conversions (inputs + all weights) ----
  struct CvtJob { int idx; h16* dst; long n; };
  const CvtJob jobs[] = {
      {0, x16, 8192L * 544}, {1, h16b, 8192L * 256},
      {3, wenc0, 128 * 64}, {5, wenc1, 128 * 32}, {7, winkqv, 384 * 128},
      {9, winf1, 256 * 128}, {11, winf2, 128 * 256}, {13, wreq, 64 * 2304},
      {15, wrkv, 256 * 128}, {17, wrq, 128 * 64}, {19, wrf1, 256 * 128},
      {21, wrf2, 64 * 256}, {23, wafk, 384 * 128}, {25, wamk, 384 * 64},
      {27, walin, 256 * 3968}, {29, wwih, 768 * 256}, {30, wwhh, 768 * 256},
      {33, walin2, 32 * 256}};
  for (const auto& j : jobs) {
    int blocks = (int)((j.n + 255) / 256);
    cvt_f2h<<<blocks, 256, 0, stream>>>((const float*)d_in[j.idx], j.dst, j.n);
  }

  // ---- encode: obs ----
  // seg0: K=64 (2 chunks)
  gemm_npre<2, 0><<<dim3(64, 1), 256, 0, stream>>>(
      x16, 544, 0, wenc0, 64, (const float*)d_in[4],
      obs, 2048, 0, 8192, 128);
  // seg1..15: K=32 (1 chunk), z-batched
  gemm_npre<1, 0><<<dim3(64, 15), 256, 0, stream>>>(
      x16 + 64, 544, 32, wenc1, 32, (const float*)d_in[6],
      obs + 128, 2048, 128, 8192, 128);

  // ---- encoder block (attention + FFN fused) -> feat ----
  enc_block<<<2048, 128, EB_W * EB_PW * 2, stream>>>(
      obs, winkqv, (const float*)d_in[8], winf1, (const float*)d_in[10],
      winf2, (const float*)d_in[12], feat);

  // ---- req = concat(hidden, feat) @ req_W^T + b  (k-outer, A read once) ----
  gemm_kouter<4><<<64, 256, 0, stream>>>(
      h16b, 256, feat, 2048, 256, 2304, wreq, 2304, (const float*)d_in[14],
      req16, 64, 8192);

  // ---- rq2 = req @ rq_W^T + b (per (b,j), broadcast over a) ----
  gemm_npre<2, 0><<<dim3(64, 1), 256, 0, stream>>>(
      req16, 64, 0, wrq, 64, (const float*)d_in[18],
      rq2, 128, 0, 8192, 128);

  // ---- response block fused -> resp_g (masked, offdiag-gathered) ----
  resp_block<<<2048, 128, RK_W * RK_PW * 2, stream>>>(
      feat, rq2, wrkv, (const float*)d_in[16], wrf1, (const float*)d_in[20],
      wrf2, (const float*)d_in[22], (const float*)d_in[2], respg);

  // ---- final attention fused -> xx (32x128 per (b,a), row 31 padded) ----
  fattn_block<<<2048, 128, FA_W * FA_PW * 2, stream>>>(
      feat, respg, wafk, (const float*)d_in[24], wamk, (const float*)d_in[26], xx);

  // ---- xl = alin(xx[:, :31, :].flatten)  (k-outer: xx read exactly once) ----
  gemm_kouter<16><<<64, 256, 0, stream>>>(
      xx, 4096, xx, 4096, 3968, 3968, walin, 3968, (const float*)d_in[28],
      xl16, 256, 8192);

  // ---- GRU gates (A-preloaded K=256, xl/h read once each) ----
  gemm_npre<8, 1><<<dim3(64, 1), 256, 0, stream>>>(
      xl16, 256, 0, wwih, 256, (const float*)d_in[31],
      gi, 768, 0, 8192, 768);
  gemm_npre<8, 1><<<dim3(64, 1), 256, 0, stream>>>(
      h16b, 256, 0, wwhh, 256, (const float*)d_in[32],
      gh, 768, 0, 8192, 768);
  gru_pw<<<(8192 * 256 + 255) / 256, 256, 0, stream>>>(
      gi, gh, (const float*)d_in[1], hnew, 8192L * 256);

  // ---- head: out = h @ alin2^T + b (f32 to d_out) ----
  gemm_npre<8, 1><<<dim3(64, 1), 256, 0, stream>>>(
      hnew, 256, 0, walin2, 256, (const float*)d_in[34],
      d_out, 32, 0, 8192, 32);
}
